// PointNetPlusPlus_13357348291184
// MI455X (gfx1250) — matrix operand order repaired and checked by reading, not compiled, not run on hardware
//
#include <hip/hip_runtime.h>
#include <hip/hip_bf16.h>
#include <math.h>

typedef __attribute__((ext_vector_type(16))) _Float16 v16h;
typedef __attribute__((ext_vector_type(8)))  float    v8f;

#define EPS_BN 1e-5f

// ---------------------------------------------------------------------------
// Fragment-packed layouts (CDNA5 ISA 7.12.2, wave32).
// A fragment (16x32 f16): per (mtile, ktile), lane/half order:
//   lanes 0-15 : halves 0-7 -> k'=0..7,  halves 8-15 -> k'=16..23   (m = lane)
//   lanes 16-31: halves 0-7 -> k'=8..15, halves 8-15 -> k'=24..31   (m = lane-16)
// B fragment (32x16 f16): lane = n%16 + 16*((k'/8)%2), half = k'%8 + 8*(k'>=16).
// GEMM loads each fragment as one contiguous 32B per-lane load.
// ---------------------------------------------------------------------------
__device__ __forceinline__ size_t apack_idx(int row, int k, int nkt) {
  int mt = row >> 4, ml = row & 15;
  int kt = k >> 5,  kk = k & 31;
  int q  = kk >> 3;                       // quarter of the 32-wide k tile
  int lane = ml + ((q & 1) << 4);
  int h = kk - (q << 3) + ((q >> 1) << 3); // q0:kk q1:kk-8 q2:kk-8 q3:kk-16
  return (((size_t)mt * nkt + kt) * 32 + lane) * 16 + h;
}

__device__ __forceinline__ size_t bpack_idx(int n, int k, int nkt) {
  int nt = n >> 4, nl = n & 15;
  int kt = k >> 5, kk = k & 31;
  int lane = nl + (((kk >> 3) & 1) << 4);
  int h = (kk & 7) + ((kk >> 4) << 3);
  return (((size_t)nt * nkt + kt) * 32 + lane) * 16 + h;
}

// ---------------------------------------------------------------------------
// GEMM: C[R x N] = A[R x K] * W[N x K]^T + bias ; packed f16 in, f32 out.
// One wave computes a 32x32 output block (2 M-tiles x 2 N-tiles, 4
// accumulators): each 4-fragment load feeds 4 WMMAs (half the bytes/FLOP of a
// 16x16 tile) and the 4 independent accumulation chains pipeline in the XDL.
// The N-edge case (head2, N=2) is hoisted out of the k-loop so the interior
// loop is a branch-free load+wmma stream.
// 4 waves per block stacked along M (block covers 128 rows).
// R is a multiple of 128, Kp a multiple of 32.
// ---------------------------------------------------------------------------
__global__ void gemm_wmma_kernel(const _Float16* __restrict__ Ap,
                                 const _Float16* __restrict__ Wp,
                                 const float* __restrict__ bias,
                                 float* __restrict__ C,
                                 int R, int nkt, int N) {
  int lane = threadIdx.x & 31;
  int wave = threadIdx.x >> 5;
  int mt0 = (blockIdx.x * 4 + wave) * 2;   // two consecutive M tiles
  int mt1 = mt0 + 1;
  int nt0 = blockIdx.y * 2;                // two consecutive N tiles
  int nt1 = nt0 + 1;
  int ntiles = (N + 15) / 16;
  bool hasN1 = nt1 < ntiles;               // wave-uniform, loop-invariant
  if (mt0 * 16 >= R) return;

  const _Float16* pa0 = Ap + ((size_t)mt0 * nkt * 32 + lane) * 16;
  const _Float16* pa1 = pa0 + (size_t)nkt * 512;
  const _Float16* pb0 = Wp + ((size_t)nt0 * nkt * 32 + lane) * 16;
  const _Float16* pb1 = pb0 + (size_t)nkt * 512;

  v8f acc00 = {}, acc01 = {}, acc10 = {}, acc11 = {};
  if (hasN1) {
    for (int kt = 0; kt < nkt; ++kt) {     // branch-free 4-WMMA body
      v16h a0 = *(const v16h*)pa0;
      v16h a1 = *(const v16h*)pa1;
      v16h b0 = *(const v16h*)pb0;
      v16h b1 = *(const v16h*)pb1;
      pa0 += 512; pa1 += 512; pb0 += 512; pb1 += 512;
      acc00 = __builtin_amdgcn_wmma_f32_16x16x32_f16(false, a0, false, b0,
                                                     (short)0, acc00, false, false);
      acc10 = __builtin_amdgcn_wmma_f32_16x16x32_f16(false, a1, false, b0,
                                                     (short)0, acc10, false, false);
      acc01 = __builtin_amdgcn_wmma_f32_16x16x32_f16(false, a0, false, b1,
                                                     (short)0, acc01, false, false);
      acc11 = __builtin_amdgcn_wmma_f32_16x16x32_f16(false, a1, false, b1,
                                                     (short)0, acc11, false, false);
    }
  } else {
    for (int kt = 0; kt < nkt; ++kt) {     // N edge: single N tile
      v16h a0 = *(const v16h*)pa0;
      v16h a1 = *(const v16h*)pa1;
      v16h b0 = *(const v16h*)pb0;
      pa0 += 512; pa1 += 512; pb0 += 512;
      acc00 = __builtin_amdgcn_wmma_f32_16x16x32_f16(false, a0, false, b0,
                                                     (short)0, acc00, false, false);
      acc10 = __builtin_amdgcn_wmma_f32_16x16x32_f16(false, a1, false, b0,
                                                     (short)0, acc10, false, false);
    }
  }

  int ncol0 = nt0 * 16 + (lane & 15);
  int ncol1 = ncol0 + 16;
  bool hi = lane >= 16;
  float bv0 = (ncol0 < N) ? bias[ncol0] : 0.0f;
  float bv1 = (hasN1 && ncol1 < N) ? bias[ncol1] : 0.0f;
#pragma unroll
  for (int r = 0; r < 8; ++r) {
    int mA = mt0 * 16 + r + (hi ? 8 : 0);
    int mB = mt1 * 16 + r + (hi ? 8 : 0);
    if (ncol0 < N) {
      C[(size_t)mA * N + ncol0] = acc00[r] + bv0;
      C[(size_t)mB * N + ncol0] = acc10[r] + bv0;
    }
    if (hasN1 && ncol1 < N) {
      C[(size_t)mA * N + ncol1] = acc01[r] + bv1;
      C[(size_t)mB * N + ncol1] = acc11[r] + bv1;
    }
  }
}

// ---------------------------------------------------------------------------
// Per-channel mean / inv-std over R rows (BatchNorm, ddof=0). Block = channel.
// ---------------------------------------------------------------------------
__global__ void stats_kernel(const float* __restrict__ X,
                             float* __restrict__ mean,
                             float* __restrict__ invstd,
                             int R, int Cc) {
  int c = blockIdx.x;
  __shared__ float ss[256];
  __shared__ float sq[256];
  float s = 0.0f, q = 0.0f;
  for (int r = threadIdx.x; r < R; r += blockDim.x) {
    float v = X[(size_t)r * Cc + c];
    s += v; q += v * v;
  }
  ss[threadIdx.x] = s; sq[threadIdx.x] = q;
  __syncthreads();
  for (int o = 128; o > 0; o >>= 1) {
    if (threadIdx.x < o) { ss[threadIdx.x] += ss[threadIdx.x + o]; sq[threadIdx.x] += sq[threadIdx.x + o]; }
    __syncthreads();
  }
  if (threadIdx.x == 0) {
    float mu = ss[0] / (float)R;
    float var = sq[0] / (float)R - mu * mu;
    mean[c] = mu;
    invstd[c] = rsqrtf(var + EPS_BN);
  }
}

// BN + ReLU; writes dense f32 (in place) and packed f16 for the next GEMM.
__global__ void bnrelu_pack_kernel(float* __restrict__ X,
                                   _Float16* __restrict__ Xp,
                                   const float* __restrict__ mean,
                                   const float* __restrict__ invstd,
                                   const float* __restrict__ g,
                                   const float* __restrict__ be,
                                   int Cc) {
  int row = blockIdx.x;
  int nkt = Cc >> 5;
  for (int c = threadIdx.x; c < Cc; c += blockDim.x) {
    size_t i = (size_t)row * Cc + c;
    float v = g[c] * (X[i] - mean[c]) * invstd[c] + be[c];
    v = v > 0.0f ? v : 0.0f;
    X[i] = v;
    Xp[apack_idx(row, c, nkt)] = (_Float16)v;
  }
}

// Pack a weight matrix W[N x K] (f32) into B-fragment order, zero padded.
__global__ void packw_kernel(const float* __restrict__ W,
                             _Float16* __restrict__ Wp,
                             int N, int K, int Np, int Kp) {
  int t = blockIdx.x * blockDim.x + threadIdx.x;
  if (t >= Np * Kp) return;
  int n = t / Kp, k = t % Kp;
  float v = (n < N && k < K) ? W[(size_t)n * K + k] : 0.0f;
  Wp[bpack_idx(n, k, Kp >> 5)] = (_Float16)v;
}

// ---------------------------------------------------------------------------
// Farthest point sampling: one block per batch; LDS distance array + argmax.
// ---------------------------------------------------------------------------
__global__ void fps_kernel(const float* __restrict__ xyz,
                           int* __restrict__ out_idx,
                           float* __restrict__ new_xyz,
                           int N, int S) {
  int b = blockIdx.x;
  const float* X = xyz + (size_t)b * N * 3;
  __shared__ float dist[4096];
  __shared__ float rv[256];
  __shared__ int   ri[256];
  __shared__ int   s_far;
  for (int i = threadIdx.x; i < N; i += blockDim.x) dist[i] = 1e10f;
  if (threadIdx.x == 0) s_far = 0;
  __syncthreads();
  for (int s = 0; s < S; ++s) {
    int far = s_far;
    if (threadIdx.x == 0) {
      out_idx[(size_t)b * S + s] = far;
      new_xyz[((size_t)b * S + s) * 3 + 0] = X[far * 3 + 0];
      new_xyz[((size_t)b * S + s) * 3 + 1] = X[far * 3 + 1];
      new_xyz[((size_t)b * S + s) * 3 + 2] = X[far * 3 + 2];
    }
    float cx = X[far * 3 + 0], cy = X[far * 3 + 1], cz = X[far * 3 + 2];
    float bv = -1.0f; int bi = 0;
    for (int i = threadIdx.x; i < N; i += blockDim.x) {
      float dx = X[i * 3 + 0] - cx;
      float dy = X[i * 3 + 1] - cy;
      float dz = X[i * 3 + 2] - cz;
      float d = dx * dx + dy * dy + dz * dz;
      float nd = fminf(dist[i], d);
      dist[i] = nd;
      if (nd > bv) { bv = nd; bi = i; }
    }
    rv[threadIdx.x] = bv; ri[threadIdx.x] = bi;
    __syncthreads();
    for (int o = 128; o > 0; o >>= 1) {
      if (threadIdx.x < o) {
        if (rv[threadIdx.x + o] > rv[threadIdx.x]) {
          rv[threadIdx.x] = rv[threadIdx.x + o];
          ri[threadIdx.x] = ri[threadIdx.x + o];
        }
      }
      __syncthreads();
    }
    if (threadIdx.x == 0) s_far = ri[0];
    __syncthreads();
  }
}

// Ball query: first k in-ball indices ascending, padded with first hit.
__global__ void ballquery_kernel(const float* __restrict__ xyz,
                                 const float* __restrict__ new_xyz,
                                 int* __restrict__ idx,
                                 int B, int N, int S, int k, float r2) {
  int t = blockIdx.x * blockDim.x + threadIdx.x;
  if (t >= B * S) return;
  int b = t / S;
  const float* nz = new_xyz + (size_t)t * 3;
  float cx = nz[0], cy = nz[1], cz = nz[2];
  int* o = idx + (size_t)t * k;
  int cnt = 0, first = 0;
  bool have = false;
  for (int j = 0; j < N && cnt < k; ++j) {
    const float* p = xyz + ((size_t)b * N + j) * 3;
    float dx = p[0] - cx, dy = p[1] - cy, dz = p[2] - cz;
    float d = dx * dx + dy * dy + dz * dz;
    if (d <= r2) {
      if (!have) { first = j; have = true; }
      o[cnt++] = j;
    }
  }
  for (; cnt < k; ++cnt) o[cnt] = first;
}

// Grouping -> packed f16 A-layout, concat(points[idx], xyz[idx]-center), zero pad.
__global__ void group_pack_kernel(const float* __restrict__ xyz,
                                  const float* __restrict__ new_xyz,
                                  const float* __restrict__ points,
                                  const int* __restrict__ idx,
                                  _Float16* __restrict__ Gp,
                                  int N, int S, int k, int Cc, int Kp) {
  int row = blockIdx.x;                 // 0 .. B*S*k-1
  int Kd = Cc + 3;
  int nkt = Kp >> 5;
  int b = row / (S * k);
  int sk = row % (S * k);
  int s = sk / k;
  int j = idx[row];
  const float* nz = new_xyz + ((size_t)b * S + s) * 3;
  for (int c = threadIdx.x; c < Kp; c += blockDim.x) {
    float v = 0.0f;
    if (c < Cc)      v = points[((size_t)b * N + j) * Cc + c];
    else if (c < Kd) v = xyz[((size_t)b * N + j) * 3 + (c - Cc)] - nz[c - Cc];
    Gp[apack_idx(row, c, nkt)] = (_Float16)v;
  }
}

// Max over k neighbors: dense f32 in -> dense f32 out @ channel offset.
__global__ void maxpool_kernel(const float* __restrict__ X,
                               float* __restrict__ out,
                               int k, int Cc, int ldOut, int co) {
  int g = blockIdx.x;
  for (int c = threadIdx.x; c < Cc; c += blockDim.x) {
    float m = -1e30f;
    for (int j = 0; j < k; ++j)
      m = fmaxf(m, X[((size_t)g * k + j) * Cc + c]);
    out[(size_t)g * ldOut + co + c] = m;
  }
}

// Dense f32 slice -> packed f16 at channel offset.
__global__ void copy_pack_kernel(const float* __restrict__ src,
                                 _Float16* __restrict__ dst,
                                 int Cs, int nkt, int doff) {
  int row = blockIdx.x;
  for (int c = threadIdx.x; c < Cs; c += blockDim.x)
    dst[apack_idx(row, doff + c, nkt)] = (_Float16)src[(size_t)row * Cs + c];
}

// Broadcast (B,C) -> packed rows = B*Npts at channel offset (fp S==1 path).
__global__ void broadcast_pack_kernel(const float* __restrict__ src,
                                      _Float16* __restrict__ dst,
                                      int Npts, int Cc, int nkt, int doff) {
  int row = blockIdx.x;
  int b = row / Npts;
  for (int c = threadIdx.x; c < Cc; c += blockDim.x)
    dst[apack_idx(row, doff + c, nkt)] = (_Float16)src[(size_t)b * Cc + c];
}

// Zero packed channels [from, to) for every row (concat K padding).
__global__ void pad_pack_kernel(_Float16* __restrict__ dst, int from, int to, int nkt) {
  int row = blockIdx.x;
  for (int c = from + threadIdx.x; c < to; c += blockDim.x)
    dst[apack_idx(row, c, nkt)] = (_Float16)0.0f;
}

// 3-NN inverse-distance interpolation -> packed f16 at channel offset.
__global__ void interp_pack_kernel(const float* __restrict__ xyz1,
                                   const float* __restrict__ xyz2,
                                   const float* __restrict__ p2,
                                   _Float16* __restrict__ dst,
                                   int B, int Npts, int S, int Cc, int nkt, int doff) {
  int t = blockIdx.x * blockDim.x + threadIdx.x;
  if (t >= B * Npts) return;
  int b = t / Npts;
  const float* q = xyz1 + (size_t)t * 3;
  float qx = q[0], qy = q[1], qz = q[2];
  float d0 = 1e30f, d1 = 1e30f, d2 = 1e30f;
  int i0 = 0, i1 = 0, i2 = 0;
  for (int s = 0; s < S; ++s) {
    const float* p = xyz2 + ((size_t)b * S + s) * 3;
    float dx = p[0] - qx, dy = p[1] - qy, dz = p[2] - qz;
    float d = dx * dx + dy * dy + dz * dz;
    if (d < d0)      { d2 = d1; i2 = i1; d1 = d0; i1 = i0; d0 = d; i0 = s; }
    else if (d < d1) { d2 = d1; i2 = i1; d1 = d; i1 = s; }
    else if (d < d2) { d2 = d; i2 = s; }
  }
  float w0 = 1.0f / (d0 + 1e-8f);
  float w1 = 1.0f / (d1 + 1e-8f);
  float w2 = 1.0f / (d2 + 1e-8f);
  float ws = w0 + w1 + w2;
  w0 /= ws; w1 /= ws; w2 /= ws;
  const float* r0 = p2 + ((size_t)b * S + i0) * Cc;
  const float* r1 = p2 + ((size_t)b * S + i1) * Cc;
  const float* r2 = p2 + ((size_t)b * S + i2) * Cc;
  for (int c = 0; c < Cc; ++c)
    dst[apack_idx(t, doff + c, nkt)] = (_Float16)(w0 * r0[c] + w1 * r1[c] + w2 * r2[c]);
}

__global__ void logsoftmax2_kernel(float* __restrict__ X, int rows) {
  int t = blockIdx.x * blockDim.x + threadIdx.x;
  if (t >= rows) return;
  float a = X[(size_t)t * 2 + 0];
  float b = X[(size_t)t * 2 + 1];
  float m = fmaxf(a, b);
  float l = m + logf(expf(a - m) + expf(b - m));
  X[(size_t)t * 2 + 0] = a - l;
  X[(size_t)t * 2 + 1] = b - l;
}

// ---------------------------------------------------------------------------
// Host orchestration
// ---------------------------------------------------------------------------
struct Layer { const float *w, *b, *g, *be; int cin, cout; _Float16* wp; int nkt; };

extern "C" void kernel_launch(void* const* d_in, const int* in_sizes, int n_in,
                              void* d_out, int out_size, void* d_ws, size_t ws_size,
                              hipStream_t stream) {
  (void)in_sizes; (void)n_in; (void)out_size; (void)ws_size;
  const int B = 16, N0 = 4096;
  const float* xyz = (const float*)d_in[0];

  // --- parse params (setup_inputs insertion order; each layer = w,b,g,be) ---
  int p = 1;
  auto nextL = [&](int cin, int cout) -> Layer {
    Layer L{ (const float*)d_in[p], (const float*)d_in[p + 1],
             (const float*)d_in[p + 2], (const float*)d_in[p + 3], cin, cout, nullptr, 0 };
    p += 4;
    return L;
  };
  Layer sa1b0[3] = { nextL(3, 32),    nextL(32, 32),   nextL(32, 64) };
  Layer sa1b1[3] = { nextL(3, 64),    nextL(64, 64),   nextL(64, 128) };
  Layer sa2b0[3] = { nextL(195, 128), nextL(128, 128), nextL(128, 256) };
  Layer sa2b1[3] = { nextL(195, 256), nextL(256, 256), nextL(256, 512) };
  Layer sa3L[2]  = { nextL(771, 512),  nextL(512, 1024) };
  Layer fp3L[2]  = { nextL(1792, 512), nextL(512, 512) };
  Layer fp2L[2]  = { nextL(704, 512),  nextL(512, 256) };
  Layer fp1L[2]  = { nextL(256, 256),  nextL(256, 128) };
  Layer head1 = nextL(128, 128);
  Layer head2{ (const float*)d_in[p], (const float*)d_in[p + 1], nullptr, nullptr, 128, 2, nullptr, 0 };

  // --- workspace arena (64B aligned) ---
  char* base = (char*)d_ws;
  size_t off = 0;
  auto allocB = [&](size_t bytes) -> void* {
    size_t a = (off + 63) & ~(size_t)63;
    void* q = base + a;
    off = a + bytes;
    return q;
  };
  auto allocF = [&](size_t n) { return (float*)allocB(n * 4); };
  auto allocH = [&](size_t n) { return (_Float16*)allocB(n * 2); };

  float* mean   = allocF(1024);
  float* invstd = allocF(1024);
  float* l1_xyz = allocF((size_t)B * 512 * 3);
  float* l2_xyz = allocF((size_t)B * 128 * 3);
  float* l1_p   = allocF((size_t)B * 512 * 192);
  float* l2_p   = allocF((size_t)B * 128 * 768);
  float* l3_p   = allocF((size_t)B * 1024);
  float* fp3o   = allocF((size_t)B * 128 * 512);
  float* fp2o   = allocF((size_t)B * 512 * 256);
  float* fp1o   = allocF((size_t)B * 4096 * 128);
  float* h1o    = allocF((size_t)B * 4096 * 128);
  _Float16* fp1o_p = allocH((size_t)B * 4096 * 128);
  _Float16* h1o_p  = allocH((size_t)B * 4096 * 128);
  int* fpsi = (int*)allocB((size_t)B * 512 * 4);
  int* bq   = (int*)allocB((size_t)B * 512 * 64 * 4);
  _Float16* Gp    = allocH((size_t)262144 * 224);        // max rows*Kp (sa2 b1)
  float*    ACT0  = allocF((size_t)262144 * 512);
  float*    ACT1  = allocF((size_t)262144 * 512);
  _Float16* ACT0p = allocH((size_t)262144 * 512);
  _Float16* ACT1p = allocH((size_t)262144 * 512);

  // --- pack all weights into B-fragment order ---
  auto prep = [&](Layer& L) {
    int Np = ((L.cout + 15) / 16) * 16;
    int Kp = ((L.cin + 31) / 32) * 32;
    L.nkt = Kp >> 5;
    L.wp = allocH((size_t)(Np / 16) * L.nkt * 512);
    int total = Np * Kp;
    packw_kernel<<<(total + 255) / 256, 256, 0, stream>>>(L.w, L.wp, L.cout, L.cin, Np, Kp);
  };
  for (int i = 0; i < 3; ++i) { prep(sa1b0[i]); prep(sa1b1[i]); prep(sa2b0[i]); prep(sa2b1[i]); }
  for (int i = 0; i < 2; ++i) { prep(sa3L[i]); prep(fp3L[i]); prep(fp2L[i]); prep(fp1L[i]); }
  prep(head1); prep(head2);

  // --- helpers ---
  auto gemm = [&](const _Float16* Ap, const Layer& L, float* Cdst, int R) {
    dim3 g(R / 128, (L.cout + 31) / 32);               // 32x32 block per wave
    gemm_wmma_kernel<<<g, dim3(128), 0, stream>>>(Ap, L.wp, L.b, Cdst, R, L.nkt, L.cout);
  };
  auto bn = [&](float* X, _Float16* Xp, const Layer& L, int R) {
    stats_kernel<<<L.cout, 256, 0, stream>>>(X, mean, invstd, R, L.cout);
    bnrelu_pack_kernel<<<R, 256, 0, stream>>>(X, Xp, mean, invstd, L.g, L.be, L.cout);
  };
  auto run_mlp = [&](const _Float16* x0p, const Layer* Ls, int nL, int R,
                     float* lastDense, _Float16* lastPack) -> float* {
    const _Float16* cur = x0p;
    float* curDense = nullptr;
    for (int i = 0; i < nL; ++i) {
      bool last = (i == nL - 1);
      float*    dstD = (last && lastDense) ? lastDense : ((curDense == ACT0) ? ACT1 : ACT0);
      _Float16* dstP = (last && lastPack)  ? lastPack  : ((cur == ACT0p) ? ACT1p : ACT0p);
      gemm(cur, Ls[i], dstD, R);
      bn(dstD, dstP, Ls[i], R);
      cur = dstP;
      curDense = dstD;
    }
    return curDense;
  };

  // ===================== SA1: N=4096 -> S=512, MSG =====================
  fps_kernel<<<B, 256, 0, stream>>>(xyz, fpsi, l1_xyz, N0, 512);
  {
    int S = 512, k = 32, R = B * S * k;              // branch 0: r=0.1, [3,32,32,64]
    ballquery_kernel<<<(B * S + 255) / 256, 256, 0, stream>>>(xyz, l1_xyz, bq, B, N0, S, k, 0.01f);
    group_pack_kernel<<<R, 64, 0, stream>>>(xyz, l1_xyz, nullptr, bq, Gp, N0, S, k, 0, 32);
    float* o = run_mlp(Gp, sa1b0, 3, R, nullptr, nullptr);
    maxpool_kernel<<<B * S, 256, 0, stream>>>(o, l1_p, k, 64, 192, 0);
  }
  {
    int S = 512, k = 64, R = B * S * k;              // branch 1: r=0.2, [3,64,64,128]
    ballquery_kernel<<<(B * S + 255) / 256, 256, 0, stream>>>(xyz, l1_xyz, bq, B, N0, S, k, 0.04f);
    group_pack_kernel<<<R, 64, 0, stream>>>(xyz, l1_xyz, nullptr, bq, Gp, N0, S, k, 0, 32);
    float* o = run_mlp(Gp, sa1b1, 3, R, nullptr, nullptr);
    maxpool_kernel<<<B * S, 256, 0, stream>>>(o, l1_p, k, 128, 192, 64);
  }

  // ===================== SA2: N=512 -> S=128, MSG ======================
  fps_kernel<<<B, 256, 0, stream>>>(l1_xyz, fpsi, l2_xyz, 512, 128);
  {
    int S = 128, k = 64, R = B * S * k;              // branch 0: r=0.2, [195,...]
    ballquery_kernel<<<(B * S + 255) / 256, 256, 0, stream>>>(l1_xyz, l2_xyz, bq, B, 512, S, k, 0.04f);
    group_pack_kernel<<<R, 256, 0, stream>>>(l1_xyz, l2_xyz, l1_p, bq, Gp, 512, S, k, 192, 224);
    float* o = run_mlp(Gp, sa2b0, 3, R, nullptr, nullptr);
    maxpool_kernel<<<B * S, 256, 0, stream>>>(o, l2_p, k, 256, 768, 0);
  }
  {
    int S = 128, k = 128, R = B * S * k;             // branch 1: r=0.4, [195,...]
    ballquery_kernel<<<(B * S + 255) / 256, 256, 0, stream>>>(l1_xyz, l2_xyz, bq, B, 512, S, k, 0.16f);
    group_pack_kernel<<<R, 256, 0, stream>>>(l1_xyz, l2_xyz, l1_p, bq, Gp, 512, S, k, 192, 224);
    float* o = run_mlp(Gp, sa2b1, 3, R, nullptr, nullptr);
    maxpool_kernel<<<B * S, 256, 0, stream>>>(o, l2_p, k, 512, 768, 256);
  }

  // ===================== SA3: group_all, [771,512,1024] ================
  {
    int R = B * 128, nkt = 800 / 32;
    copy_pack_kernel<<<R, 64, 0, stream>>>(l2_xyz, Gp, 3, nkt, 0);       // xyz first
    copy_pack_kernel<<<R, 256, 0, stream>>>(l2_p, Gp, 768, nkt, 3);      // then points
    pad_pack_kernel<<<R, 32, 0, stream>>>(Gp, 771, 800, nkt);
    float* o = run_mlp(Gp, sa3L, 2, R, nullptr, nullptr);
    maxpool_kernel<<<B, 256, 0, stream>>>(o, l3_p, 128, 1024, 1024, 0);  // max over 128 pts
  }

  // ===================== FP3: S==1 broadcast, [1792,512,512] ===========
  {
    int R = B * 128, nkt = 1792 / 32;
    copy_pack_kernel<<<R, 256, 0, stream>>>(l2_p, Gp, 768, nkt, 0);
    broadcast_pack_kernel<<<R, 256, 0, stream>>>(l3_p, Gp, 128, 1024, nkt, 768);
    run_mlp(Gp, fp3L, 2, R, fp3o, nullptr);
  }

  // ===================== FP2: 3-NN interp, [704,512,256] ===============
  {
    int R = B * 512, nkt = 704 / 32;
    copy_pack_kernel<<<R, 256, 0, stream>>>(l1_p, Gp, 192, nkt, 0);
    interp_pack_kernel<<<(R + 255) / 256, 256, 0, stream>>>(l1_xyz, l2_xyz, fp3o, Gp, B, 512, 128, 512, nkt, 192);
    run_mlp(Gp, fp2L, 2, R, fp2o, nullptr);
  }

  // ===================== FP1: 3-NN interp, [256,256,128] ===============
  {
    int R = B * 4096, nkt = 256 / 32;
    interp_pack_kernel<<<(R + 255) / 256, 256, 0, stream>>>(xyz, l1_xyz, fp2o, Gp, B, 4096, 512, 256, nkt, 0);
    run_mlp(Gp, fp1L, 2, R, fp1o, fp1o_p);
  }

  // ===================== Head ==========================================
  {
    int R = B * 4096;
    gemm(fp1o_p, head1, h1o, R);
    bn(h1o, h1o_p, head1, R);
    gemm(h1o_p, head2, (float*)d_out, R);            // 128 -> 2, bias only
    logsoftmax2_kernel<<<(R + 255) / 256, 256, 0, stream>>>((float*)d_out, R);
  }
}
